// MyLoss_v2_20684562497963
// MI455X (gfx1250) — compile-verified, matrix operand order checked by reading, not run on hardware
//
#include <hip/hip_runtime.h>
#include <math.h>

// ---------------- problem constants (fixed by reference setup_inputs) -------
#define NB   8
#define NA   3
#define GSZ  128
#define MGT  64
#define NPI  (NA*GSZ*GSZ)      // 49152 preds per image
#define NTOT (NB*NPI)          // 393216 preds total
#define NBLK (NTOT/256)        // 1536 reduction blocks
#define NEGF     (-1e9f)
#define NEGHALF  (-5e8f)

typedef __attribute__((ext_vector_type(8))) int   v8i;
typedef __attribute__((ext_vector_type(8))) float v8f;
typedef __attribute__((ext_vector_type(2))) float v2f;

// GIoU exactly as the reference (EPS = 1e-16, clip at 0)
__device__ __forceinline__ void iou_giou(float cx, float cy, float w, float h,
                                         float gx1f, float gy1f, float gx2f, float gy2f,
                                         float& iou, float& gou) {
  float px1 = cx - 0.5f*w, px2 = cx + 0.5f*w;
  float py1 = cy - 0.5f*h, py2 = cy + 0.5f*h;
  float iw = fmaxf(fminf(px2, gx2f) - fmaxf(px1, gx1f), 0.0f);
  float ih = fmaxf(fminf(py2, gy2f) - fmaxf(py1, gy1f), 0.0f);
  float inter = iw * ih;
  float ap = (px2 - px1) * (py2 - py1);
  float ag = (gx2f - gx1f) * (gy2f - gy1f);
  float un = ap + ag - inter;
  iou = inter / (un + 1e-16f);
  float cw = fmaxf(px2, gx2f) - fminf(px1, gx1f);
  float ch = fmaxf(py2, gy2f) - fminf(py1, gy1f);
  float ac = cw * ch;
  gou = iou - (ac - un) / (ac + 1e-16f);
}

// ---------------- kernel 0: GT preprocessing --------------------------------
__global__ void k_prep(const float* __restrict__ gts,
                       float* gx1, float* gy1, float* gx2, float* gy2,
                       int* gtb, int* has_gt) {
  int m = threadIdx.x;
  if (m < MGT) {
    float cx = gts[m*5+1], cy = gts[m*5+2], w = gts[m*5+3], h = gts[m*5+4];
    gx1[m] = cx - 0.5f*w; gx2[m] = cx + 0.5f*w;
    gy1[m] = cy - 0.5f*h; gy2[m] = cy + 0.5f*h;
    gtb[m] = (int)gts[m*5+0];
  }
  if (m < NB) {
    int any = 0;
    for (int j = 0; j < MGT; ++j) any |= ((int)gts[j*5+0] == m) ? 1 : 0;
    has_gt[m] = any;
  }
}

// ---------------- kernel A: decode predictions ------------------------------
__global__ void __launch_bounds__(256) k_pred(const float* __restrict__ outp,
                                              float4* pd4, float* conf,
                                              int* tp, int* pidx) {
  int p = blockIdx.x * 256 + threadIdx.x;
  int b = p / NPI;
  int i = p - b * NPI;
  int a = i / (GSZ*GSZ);
  int r = i - a * (GSZ*GSZ);
  int y = r / GSZ;
  int x = r - y * GSZ;
  const float* base = outp + ((((size_t)b*(NA*5) + (size_t)a*5) * GSZ + y) * GSZ + x);
  const size_t S = (size_t)GSZ * GSZ;
  float t0 = base[0];
  float t1 = base[S];
  float t2 = base[2*S];
  float t3 = base[3*S];
  float t4 = base[4*S];
  float xs = (tanhf(t0) + 0.5f + (float)x) * (1.0f/(float)GSZ);
  float ys = (tanhf(t1) + 0.5f + (float)y) * (1.0f/(float)GSZ);
  float wv = expf(-t2*t2);
  float hv = expf(-t3*t3);
  float cf = 1.0f / (1.0f + expf(-t4));
  pd4[p]  = make_float4(xs, ys, wv, hv);
  conf[p] = cf;
  tp[p]   = 0;
  pidx[p] = 0;
}

// ---------------- kernel B: greedy GIoU matching (1 block per image) --------
// Pred-major scan: each thread reads a pred once per chunk of <=8 active GTs,
// so iteration-1 L2 traffic is 8x lower than a wave-per-GT scan.
__global__ void __launch_bounds__(1024) k_match(const float4* __restrict__ pd4,
                const float* __restrict__ gx1, const float* __restrict__ gy1,
                const float* __restrict__ gx2, const float* __restrict__ gy2,
                const int* __restrict__ gtb, int* tp, int* pidx) {
  __shared__ unsigned int bits[NPI/32];             // matched-pred bitset (6KB)
  __shared__ float sx1[MGT], sy1[MGT], sx2[MGT], sy2[MGT];
  __shared__ float valj[MGT];
  __shared__ int   pdj[MGT];
  __shared__ int   sel[MGT], validm[MGT];
  __shared__ int   actlist[MGT];
  __shared__ int   nact;
  __shared__ float redv[32][8];
  __shared__ int   redi[32][8];

  const int b   = blockIdx.x;
  const int tid = threadIdx.x;
  const int wid = tid >> 5, lane = tid & 31;

  for (int j = tid; j < NPI/32; j += 1024) bits[j] = 0u;
  if (tid < MGT) {
    sx1[tid] = gx1[tid]; sy1[tid] = gy1[tid];
    sx2[tid] = gx2[tid]; sy2[tid] = gy2[tid];
    validm[tid] = (gtb[tid] == b) ? 1 : 0;
    sel[tid] = 0;
  }
  __syncthreads();

  const float4* pdb = pd4 + (size_t)b * NPI;

  for (int iter = 0; iter < MGT; ++iter) {
    if (tid == 0) {                       // compact list of active GTs
      int c = 0;
      for (int m = 0; m < MGT; ++m) if (validm[m] && !sel[m]) actlist[c++] = m;
      nact = c;
    }
    if (tid < MGT) { valj[tid] = NEGF; pdj[tid] = 0; }
    __syncthreads();
    const int na = nact;
    if (na == 0) break;

    for (int c0 = 0; c0 < na; c0 += 8) {
      const int nc = (na - c0 < 8) ? (na - c0) : 8;
      float bv[8]; int bi[8];
      float a1[8], b1[8], a2[8], b2[8];
      #pragma unroll
      for (int j = 0; j < 8; ++j) {       // pad with harmless zero boxes
        bv[j] = NEGF; bi[j] = 0;
        a1[j] = 0.f; b1[j] = 0.f; a2[j] = 0.f; b2[j] = 0.f;
      }
      for (int j = 0; j < nc; ++j) {
        int m = actlist[c0 + j];
        a1[j] = sx1[m]; b1[j] = sy1[m]; a2[j] = sx2[m]; b2[j] = sy2[m];
      }
      // one pass over all preds, 8 GTs evaluated per pred load
      for (int i = tid; i < NPI; i += 1024) {
        if (!(bits[i >> 5] & (1u << (i & 31)))) {
          float4 pb = pdb[i];
          #pragma unroll
          for (int j = 0; j < 8; ++j) {
            float io, go;
            iou_giou(pb.x, pb.y, pb.z, pb.w, a1[j], b1[j], a2[j], b2[j], io, go);
            if (go > bv[j]) { bv[j] = go; bi[j] = i; }
          }
        }
      }
      // wave-level reduce per GT (first-max / min-index tie rule)
      for (int j = 0; j < nc; ++j) {
        float v = bv[j]; int ix = bi[j];
        #pragma unroll
        for (int off = 16; off > 0; off >>= 1) {
          float ov = __shfl_xor(v, off, 32);
          int   oi = __shfl_xor(ix, off, 32);
          if (ov > v || (ov == v && oi < ix)) { v = ov; ix = oi; }
        }
        if (lane == 0) { redv[wid][j] = v; redi[wid][j] = ix; }
      }
      __syncthreads();
      // cross-wave reduce: wave j reduces the 32 wave-results for GT j
      if (wid < nc) {
        float v = redv[lane][wid]; int ix = redi[lane][wid];
        #pragma unroll
        for (int off = 16; off > 0; off >>= 1) {
          float ov = __shfl_xor(v, off, 32);
          int   oi = __shfl_xor(ix, off, 32);
          if (ov > v || (ov == v && oi < ix)) { v = ov; ix = oi; }
        }
        if (lane == 0) { int m = actlist[c0 + wid]; valj[m] = v; pdj[m] = ix; }
      }
      __syncthreads();
    }

    // conflict resolution: GT with highest val (lowest index on ties) wins pred
    if (tid == 0) {
      for (int m = 0; m < MGT; ++m) {
        if (valj[m] <= NEGHALF) continue;
        int i = pdj[m];
        float best = NEGF; int bestm = -1;
        for (int mm = 0; mm < MGT; ++mm)
          if (pdj[mm] == i && valj[mm] > best) { best = valj[mm]; bestm = mm; }
        if (bestm == m) {
          bits[i >> 5] |= 1u << (i & 31);
          sel[m] = 1;
          tp[(size_t)b * NPI + i]   = 1;
          pidx[(size_t)b * NPI + i] = m;
        }
      }
    }
    __syncthreads();
  }
}

// ---------------- kernel C: fused per-pred loss partials --------------------
// slots: 0 xyx 1 xyy 2 whw 3 whh 4 gou 5 iou 6 bce 7 inter 8 pdarea
//        9 tpcnt 10 igcnt 11 tp&ig cnt (WMMA IU8)
__global__ void __launch_bounds__(256) k_loss(const float4* __restrict__ pd4,
      const float* __restrict__ conf, const int* __restrict__ tp,
      const int* __restrict__ pidx,
      const float* __restrict__ gx1, const float* __restrict__ gy1,
      const float* __restrict__ gx2, const float* __restrict__ gy2,
      const int* __restrict__ gtb, const int* __restrict__ has_gt,
      const float* __restrict__ gts, float* __restrict__ partial) {
  __shared__ float sx1[MGT], sy1[MGT], sx2[MGT], sy2[MGT];
  __shared__ float sgt[MGT*4];
  __shared__ int   sb[MGT];
  __shared__ float wsum[8][9];
  __shared__ unsigned char flagb[1024];

  const int tid = threadIdx.x;
  if (tid < MGT) {
    sx1[tid] = gx1[tid]; sy1[tid] = gy1[tid];
    sx2[tid] = gx2[tid]; sy2[tid] = gy2[tid];
    sb[tid]  = gtb[tid];
    sgt[tid*4+0] = gts[tid*5+1]; sgt[tid*4+1] = gts[tid*5+2];
    sgt[tid*4+2] = gts[tid*5+3]; sgt[tid*4+3] = gts[tid*5+4];
  }
  __syncthreads();

  const int p = blockIdx.x * 256 + tid;
  const int b = p / NPI;
  float4 pb = pd4[p];
  float  cf = conf[p];
  int   tpi = tp[p];
  int    hg = has_gt[b];

  int pi = 0;
  if (tpi) {
    pi = pidx[p];
  } else if (hg) {                      // best valid GT by GIoU (first-max)
    float bv = NEGF;
    for (int m = 0; m < MGT; ++m) {
      if (sb[m] == b) {
        float io, go;
        iou_giou(pb.x, pb.y, pb.z, pb.w, sx1[m], sy1[m], sx2[m], sy2[m], io, go);
        if (go > bv) { bv = go; pi = m; }
      }
    }
  }
  float piou = 0.f, pgou = 0.f;
  if (hg) iou_giou(pb.x, pb.y, pb.z, pb.w, sx1[pi], sy1[pi], sx2[pi], sy2[pi], piou, pgou);

  float tpf = tpi ? 1.f : 0.f;
  int   ig  = (((piou > 0.5f) ? 0 : 1) | tpi) ? 1 : 0;
  float igf = (float)ig;
  float tx = sgt[pi*4+0], ty = sgt[pi*4+1], tw = sgt[pi*4+2], th = sgt[pi*4+3];
  float c  = fminf(fmaxf(cf, 1e-7f), 1.f - 1e-7f);

  float acc[9];
  acc[0] = (pb.x - tx) * (pb.x - tx) * tpf;
  acc[1] = (pb.y - ty) * (pb.y - ty) * tpf;
  float lw = logf(pb.z) - logf(tw);
  float lh = logf(pb.w) - logf(th);
  acc[2] = lw * lw * tpf;
  acc[3] = lh * lh * tpf;
  acc[4] = pgou * tpf;
  acc[5] = piou * tpf;
  acc[6] = -(tpf * logf(c) + (1.f - tpf) * log1pf(-c)) * igf;
  acc[7] = c * tpf * igf;
  acc[8] = c * c * igf;

  const int wid = tid >> 5, lane = tid & 31;
  #pragma unroll
  for (int k = 0; k < 9; ++k) {
    float v = acc[k];
    #pragma unroll
    for (int off = 16; off > 0; off >>= 1) v += __shfl_xor(v, off, 32);
    if (lane == 0) wsum[wid][k] = v;
  }
  // integer flags as bytes: rows 0-3 = tp, 4-7 = ig, 8-11 = tp&ig, 12-15 = 0
  flagb[tid]       = (unsigned char)tpi;
  flagb[256 + tid] = (unsigned char)ig;
  flagb[512 + tid] = (unsigned char)(tpi & ig);
  flagb[768 + tid] = 0;
  __syncthreads();

  float* pout = partial + (size_t)blockIdx.x * 16;
  if (tid < 9) {
    float s = 0.f;
    for (int w = 0; w < 8; ++w) s += wsum[w][tid];
    pout[tid] = s;
  }
  if (wid == 0) {
    // 8-bit A (16x64) per ISA layout; with B = all-ones, D[m][n] = row-sum.
    int mrow = lane & 15;
    int hi8  = (lane >= 16) ? 8 : 0;
    v8i A;
    #pragma unroll
    for (int v = 0; v < 8; ++v) {
      int kbase = (v >> 1) * 16;
      int klo   = ((v & 1) << 2) + hi8;
      unsigned int word = 0;
      #pragma unroll
      for (int bb = 0; bb < 4; ++bb)
        word |= ((unsigned int)flagb[mrow * 64 + kbase + klo + bb]) << (8 * bb);
      A[v] = (int)word;
    }
    v8i B, C;
    #pragma unroll
    for (int v = 0; v < 8; ++v) { B[v] = 0x01010101; C[v] = 0; }
    v8i D = __builtin_amdgcn_wmma_i32_16x16x64_iu8(false, A, false, B, C, false, false);
    // D layout: lane0 holds M=0..7 at N=0, lane16 holds M=8..15 at N=0
    if (lane == 0) {
      pout[9]  = (float)(D[0] + D[1] + D[2] + D[3]);   // tp count
      pout[10] = (float)(D[4] + D[5] + D[6] + D[7]);   // ignore count
      pout[12] = 0.f; pout[13] = 0.f; pout[14] = 0.f; pout[15] = 0.f;
    }
    if (lane == 16) {
      pout[11] = (float)(D[0] + D[1] + D[2] + D[3]);   // tp&ig count (gt_area)
    }
  }
}

// ---------------- kernel D: deterministic final reduction -------------------
__global__ void k_final(const float* __restrict__ partial, float* __restrict__ outv) {
  __shared__ float tot[16];
  int tid = threadIdx.x;          // 32 threads, one wave
#if defined(__gfx1250__) && __has_builtin(__builtin_amdgcn_wmma_f32_16x16x4_f32)
  {
    // D += ones(16x4) x B(4x16), 384 K-chunks of 4 partial rows.
    // A all-ones => D[m][n] = sum over all k of B[k][n]; K placement is
    // irrelevant, only lane->N (= lane & 15) matters. Full f32 precision.
    v8f acc = {};
    v2f aones; aones[0] = 1.0f; aones[1] = 1.0f;
    const int rbase = (tid < 16) ? 0 : 2;   // B rows held by this lane half
    const int n     = tid & 15;
    for (int c = 0; c < NBLK/4; ++c) {
      v2f bm;
      bm[0] = partial[(size_t)(c*4 + rbase + 0) * 16 + n];
      bm[1] = partial[(size_t)(c*4 + rbase + 1) * 16 + n];
      acc = __builtin_amdgcn_wmma_f32_16x16x4_f32(false, aones, false, bm,
                                                  (short)0, acc, false, false);
    }
    // all D rows identical; lane l<16 holds N=l, VGPR0 -> M=0
    if (tid < 16) tot[tid] = acc[0];
  }
  __syncthreads();
#else
  if (tid < 12) {
    float s = 0.f;
    for (int bk = 0; bk < NBLK; ++bk) s += partial[(size_t)bk * 16 + tid];
    tot[tid] = s;
  }
  __syncthreads();
#endif
  if (tid == 0) {
    float cnt      = fmaxf(tot[9], 1.f);
    float loss_xy  = (tot[0] + tot[1]) / cnt;
    float loss_wh  = (tot[2] + tot[3]) / cnt;
    float loss_gou = 1.f - tot[4] / cnt;
    float loss_iou = 1.f - tot[5] / cnt;
    float icnt     = fmaxf(tot[10], 1.f);
    float loss_bce = tot[6] / icnt;
    float loss_dice = 1.f - (2.f * tot[7] + 1.f) / (tot[8] + tot[11] + 1.f);
    float loss_conf = loss_dice + loss_bce;
    float total = loss_conf + 2.f * loss_gou + loss_wh + loss_xy;
    outv[0] = loss_xy; outv[1] = loss_wh; outv[2] = loss_conf;
    outv[3] = loss_iou; outv[4] = loss_gou; outv[5] = total;
  }
}

// ---------------- host launch ----------------------------------------------
extern "C" void kernel_launch(void* const* d_in, const int* in_sizes, int n_in,
                              void* d_out, int out_size, void* d_ws, size_t ws_size,
                              hipStream_t stream) {
  (void)in_sizes; (void)n_in; (void)out_size; (void)ws_size;
  const float* out_in = (const float*)d_in[0];  // [8,15,128,128]
  const float* gts_in = (const float*)d_in[1];  // [64,5]

  char* ws = (char*)d_ws;
  size_t off = 0;
  float4* pd4  = (float4*)(ws + off); off += (size_t)NTOT * 16;
  float*  conf = (float*) (ws + off); off += (size_t)NTOT * 4;
  int*    tp   = (int*)   (ws + off); off += (size_t)NTOT * 4;
  int*    pidx = (int*)   (ws + off); off += (size_t)NTOT * 4;
  float*  gx1  = (float*) (ws + off); off += MGT * 4;
  float*  gy1  = (float*) (ws + off); off += MGT * 4;
  float*  gx2  = (float*) (ws + off); off += MGT * 4;
  float*  gy2  = (float*) (ws + off); off += MGT * 4;
  int*    gtb  = (int*)   (ws + off); off += MGT * 4;
  int*    hasg = (int*)   (ws + off); off += 16 * 4;       // 8 used, pad to 16B
  float*  partial = (float*)(ws + off); off += (size_t)NBLK * 16 * 4;

  k_prep <<<1, 64, 0, stream>>>(gts_in, gx1, gy1, gx2, gy2, gtb, hasg);
  k_pred <<<NTOT/256, 256, 0, stream>>>(out_in, pd4, conf, tp, pidx);
  k_match<<<NB, 1024, 0, stream>>>(pd4, gx1, gy1, gx2, gy2, gtb, tp, pidx);
  k_loss <<<NTOT/256, 256, 0, stream>>>(pd4, conf, tp, pidx,
                                        gx1, gy1, gx2, gy2, gtb, hasg,
                                        gts_in, partial);
  k_final<<<1, 32, 0, stream>>>(partial, (float*)d_out);
}
